// ShapeEncoder_11235634446349
// MI455X (gfx1250) — compile-verified, matrix operand order checked by reading, not run on hardware
//
#include <hip/hip_runtime.h>
#include <math.h>

#define NNODES 20000
#define NEDGES 320000
#define ETOT   (NEDGES + NNODES)
#define HCDIM  512
#define NGRAPH 16
#define EPS_SM 1e-16f

typedef __attribute__((ext_vector_type(16))) __bf16 v16bf;
typedef __attribute__((ext_vector_type(8)))  float  v8f;

union V16 {
  v16bf v;
  uint4 q[2];
};

__device__ __forceinline__ unsigned short f32_to_bf16(float f) {
  unsigned int u = __float_as_uint(f);
  u += 0x7fffu + ((u >> 16) & 1u);   // round-to-nearest-even
  return (unsigned short)(u >> 16);
}

// Standard float atomic-max via monotonic int punning (values init'd to -inf).
__device__ __forceinline__ void atomicMaxF32(float* addr, float val) {
  if (!(val < 0.0f)) atomicMax((int*)addr, __float_as_int(val));
  else               atomicMin((unsigned int*)addr, (unsigned int)__float_as_int(val));
}

// ---------------------------------------------------------------------------
// Generic fill
// ---------------------------------------------------------------------------
__global__ void __launch_bounds__(256) k_fill(float* p, float v, int n) {
  int i = blockIdx.x * 256 + threadIdx.x;
  if (i < n) p[i] = v;
}

// ---------------------------------------------------------------------------
// Weight convert+transpose: Wt[n*K + k] = bf16(W[k*N + n])
// ---------------------------------------------------------------------------
__global__ void __launch_bounds__(256)
k_w_to_bf16t(const float* __restrict__ W, unsigned short* __restrict__ Wt, int K, int N) {
  int i = blockIdx.x * 256 + threadIdx.x;
  if (i >= K * N) return;
  int k = i / N, n = i - k * N;
  Wt[(size_t)n * K + k] = f32_to_bf16(W[i]);
}

// ---------------------------------------------------------------------------
// Encoder layer 1: h1 = relu(x @ W1 + b1), IN_DIM=3 -> 128, emit bf16
// ---------------------------------------------------------------------------
__global__ void __launch_bounds__(256)
k_encoder1(const float* __restrict__ x, const float* __restrict__ W1,
           const float* __restrict__ b1, unsigned short* __restrict__ h1bf) {
  int i = blockIdx.x * 256 + threadIdx.x;
  if (i >= NNODES * 128) return;
  int m = i >> 7, j = i & 127;
  float s = b1[j] + x[m * 3 + 0] * W1[0 * 128 + j]
                  + x[m * 3 + 1] * W1[1 * 128 + j]
                  + x[m * 3 + 2] * W1[2 * 128 + j];
  h1bf[i] = f32_to_bf16(fmaxf(s, 0.0f));
}

// ---------------------------------------------------------------------------
// WMMA GEMM: C[M,N] = A[M,K](bf16) * Bt[N,K](bf16)^T + bias, f32 out,
// optional bf16 mirror. 8 waves/block, one 16x16 tile per wave,
// v_wmma_f32_16x16x32_bf16 over K in steps of 32.
// Fragment layout per CDNA5 ISA 7.12.2:
//   A (16-bit, 16x32): lanes 0-15 hold M=lane, K = [0..7] then [16..23]
//                      lanes 16-31 hold M=lane-16, K = [8..15] then [24..31]
//   B mirrors A with N in place of M (Bt is N-major so loads are contiguous).
//   C/D: VGPR r -> row m0+r (lanes 0-15) / m0+8+r (lanes 16-31), col n0+(lane&15)
// ---------------------------------------------------------------------------
__global__ void __launch_bounds__(256)
k_wmma_gemm(const unsigned short* __restrict__ A,
            const unsigned short* __restrict__ Bt,
            const float* __restrict__ bias,
            float* __restrict__ C,
            unsigned short* __restrict__ Cbf,
            int M, int N, int K, int ldc) {
  const int wave = threadIdx.x >> 5;
  const int lane = threadIdx.x & 31;
  const int m0 = (blockIdx.x * 2 + (wave >> 2)) * 16;
  const int n0 = (blockIdx.y * 4 + (wave & 3)) * 16;
  if (m0 >= M || n0 >= N) return;               // wave-uniform: EXEC stays all-1s
  const int half = lane >> 4;
  const int l15  = lane & 15;
  const int kb   = half * 8;
  const unsigned short* arow = A  + (size_t)(m0 + l15) * K;
  const unsigned short* brow = Bt + (size_t)(n0 + l15) * K;
  v8f acc = {};
  for (int k = 0; k < K; k += 32) {
    V16 a, b;
    a.q[0] = *(const uint4*)(arow + k + kb);
    a.q[1] = *(const uint4*)(arow + k + kb + 16);
    b.q[0] = *(const uint4*)(brow + k + kb);
    b.q[1] = *(const uint4*)(brow + k + kb + 16);
    acc = __builtin_amdgcn_wmma_f32_16x16x32_bf16(false, a.v, false, b.v,
                                                  (short)0, acc, false, false);
  }
  const int col   = n0 + l15;
  const int rbase = m0 + half * 8;
  const float bv  = bias ? bias[col] : 0.0f;
#pragma unroll
  for (int r = 0; r < 8; ++r) {
    float v = acc[r] + bv;
    size_t o = (size_t)(rbase + r) * ldc + col;
    C[o] = v;
    if (Cbf) Cbf[o] = f32_to_bf16(v);
  }
}

// ---------------------------------------------------------------------------
// Edge pass 1: logit[e][h] = sum_c att[h,c]*leaky_relu(xl[src]+xr[dst]);
// atomic segment-max into mmax[dst][h]. One wave per edge, 16 ch/lane.
// ---------------------------------------------------------------------------
__global__ void __launch_bounds__(256)
k_edge_logit(const int* __restrict__ ei, const float* __restrict__ xl,
             const float* __restrict__ xr, const float* __restrict__ att,
             float* __restrict__ logit, float* __restrict__ mmax) {
  int wid  = (blockIdx.x * 256 + threadIdx.x) >> 5;
  int lane = threadIdx.x & 31;
  if (wid >= ETOT) return;
  int src, dst;
  if (wid < NEDGES) { src = ei[wid]; dst = ei[NEDGES + wid]; }
  else              { src = wid - NEDGES; dst = src; }
  const float* pl = xl + (size_t)src * HCDIM + lane * 16;
  const float* pr = xr + (size_t)dst * HCDIM + lane * 16;
  const float* pa = att + lane * 16;   // channel c maps to att[h*128 + c%128] == att[c]
  float s = 0.0f;
#pragma unroll
  for (int i = 0; i < 16; i += 4) {
    float4 a = *(const float4*)(pl + i);
    float4 b = *(const float4*)(pr + i);
    float4 w = *(const float4*)(pa + i);
    float e;
    e = a.x + b.x; s += w.x * (e > 0.f ? e : 0.2f * e);
    e = a.y + b.y; s += w.y * (e > 0.f ? e : 0.2f * e);
    e = a.z + b.z; s += w.z * (e > 0.f ? e : 0.2f * e);
    e = a.w + b.w; s += w.w * (e > 0.f ? e : 0.2f * e);
  }
  // 8 lanes per head (128 ch): reduce within group of 8
  s += __shfl_xor(s, 1); s += __shfl_xor(s, 2); s += __shfl_xor(s, 4);
  if ((lane & 7) == 0) {
    int h = lane >> 3;
    logit[(size_t)wid * 4 + h] = s;
    atomicMaxF32(&mmax[dst * 4 + h], s);
  }
}

// ---------------------------------------------------------------------------
// Edge pass 2: ex = exp(logit - m[dst]); den[dst] += ex. Thread per (e,h).
// ---------------------------------------------------------------------------
__global__ void __launch_bounds__(256)
k_edge_exp(const int* __restrict__ ei, const float* __restrict__ logit,
           const float* __restrict__ mmax, float* __restrict__ ex,
           float* __restrict__ den) {
  int i = blockIdx.x * 256 + threadIdx.x;
  if (i >= ETOT * 4) return;
  int e = i >> 2, h = i & 3;
  int dst = (e < NEDGES) ? ei[NEDGES + e] : e - NEDGES;
  float m = mmax[dst * 4 + h];
  if (!(m > -1e37f && m < 1e37f)) m = 0.0f;    // mirror reference isfinite clamp
  float v = __expf(logit[i] - m);
  ex[i] = v;
  atomicAdd(&den[dst * 4 + h], v);
}

// ---------------------------------------------------------------------------
// Edge pass 3: accum[dst] += (ex/den[dst]) * xl[src]. Wave per edge.
// ---------------------------------------------------------------------------
__global__ void __launch_bounds__(256)
k_edge_scatter(const int* __restrict__ ei, const float* __restrict__ xl,
               const float* __restrict__ ex, const float* __restrict__ den,
               float* __restrict__ accum) {
  int wid  = (blockIdx.x * 256 + threadIdx.x) >> 5;
  int lane = threadIdx.x & 31;
  if (wid >= ETOT) return;
  int src, dst;
  if (wid < NEDGES) { src = ei[wid]; dst = ei[NEDGES + wid]; }
  else              { src = wid - NEDGES; dst = src; }
  int h = lane >> 3;
  float alpha = ex[(size_t)wid * 4 + h] / (den[dst * 4 + h] + EPS_SM);
  const float* pl = xl + (size_t)src * HCDIM + lane * 16;
  float*       po = accum + (size_t)dst * HCDIM + lane * 16;
#pragma unroll
  for (int i = 0; i < 16; ++i) atomicAdd(po + i, alpha * pl[i]);
}

// ---------------------------------------------------------------------------
// Node epilogue: h' = elu(layer_norm(accum+bias)*g+b + resid); emit f32+bf16.
// Wave per node, wave32 shfl reductions for mean/var.
// ---------------------------------------------------------------------------
__global__ void __launch_bounds__(256)
k_node_epilogue(const float* __restrict__ accum, const float* __restrict__ bias,
                const float* __restrict__ ln_g, const float* __restrict__ ln_b,
                const float* __restrict__ resid, float* __restrict__ hout,
                unsigned short* __restrict__ hout_bf) {
  int n    = (blockIdx.x * 256 + threadIdx.x) >> 5;
  int lane = threadIdx.x & 31;
  if (n >= NNODES) return;
  int c0 = lane * 16;
  const float* p = accum + (size_t)n * HCDIM + c0;
  float v[16], sum = 0.f, ssq = 0.f;
#pragma unroll
  for (int i = 0; i < 16; ++i) {
    float t = p[i] + bias[c0 + i];
    v[i] = t; sum += t; ssq += t * t;
  }
#pragma unroll
  for (int m = 1; m < 32; m <<= 1) { sum += __shfl_xor(sum, m); ssq += __shfl_xor(ssq, m); }
  float mu  = sum * (1.0f / HCDIM);
  float var = ssq * (1.0f / HCDIM) - mu * mu;
  float inv = rsqrtf(var + 1e-5f);
  const float* pr = resid + (size_t)n * HCDIM + c0;
#pragma unroll
  for (int i = 0; i < 16; ++i) {
    float y = (v[i] - mu) * inv * ln_g[c0 + i] + ln_b[c0 + i] + pr[i];
    y = (y > 0.f) ? y : expm1f(y);               // ELU(alpha=1)
    size_t o = (size_t)n * HCDIM + c0 + i;
    hout[o]    = y;
    hout_bf[o] = f32_to_bf16(y);
  }
}

// ---------------------------------------------------------------------------
// Gate pooling
// ---------------------------------------------------------------------------
__global__ void __launch_bounds__(256)
k_gate_logit(const float* __restrict__ h, const float* __restrict__ gw,
             const float* __restrict__ gb, const int* __restrict__ batch,
             float* __restrict__ gate, float* __restrict__ gm) {
  int n    = (blockIdx.x * 256 + threadIdx.x) >> 5;
  int lane = threadIdx.x & 31;
  if (n >= NNODES) return;
  const float* p = h + (size_t)n * HCDIM + lane * 16;
  const float* w = gw + lane * 16;
  float s = 0.f;
#pragma unroll
  for (int i = 0; i < 16; ++i) s += p[i] * w[i];
#pragma unroll
  for (int m = 1; m < 32; m <<= 1) s += __shfl_xor(s, m);
  if (lane == 0) {
    s += gb[0];
    gate[n] = s;
    atomicMaxF32(&gm[batch[n]], s);
  }
}

__global__ void __launch_bounds__(256)
k_gate_exp(const float* __restrict__ gate, const float* __restrict__ gm,
           const int* __restrict__ batch, float* __restrict__ ge,
           float* __restrict__ gden) {
  int n = blockIdx.x * 256 + threadIdx.x;
  if (n >= NNODES) return;
  int b = batch[n];
  float m = gm[b];
  if (!(m > -1e37f && m < 1e37f)) m = 0.f;
  float v = __expf(gate[n] - m);
  ge[n] = v;
  atomicAdd(&gden[b], v);
}

__global__ void __launch_bounds__(256)
k_gate_pool(const float* __restrict__ h, const float* __restrict__ ge,
            const float* __restrict__ gden, const int* __restrict__ batch,
            float* __restrict__ pooled) {
  int n    = (blockIdx.x * 256 + threadIdx.x) >> 5;
  int lane = threadIdx.x & 31;
  if (n >= NNODES) return;
  int b = batch[n];
  float w = ge[n] / (gden[b] + EPS_SM);
  const float* p  = h + (size_t)n * HCDIM + lane * 16;
  float*       po = pooled + (size_t)b * HCDIM + lane * 16;
#pragma unroll
  for (int i = 0; i < 16; ++i) atomicAdd(po + i, w * p[i]);
}

// ---------------------------------------------------------------------------
// Final projection + L2 normalize. Tiny (16 rows): single block, LDS staging.
// ---------------------------------------------------------------------------
__global__ void __launch_bounds__(256)
k_proj(const float* __restrict__ pooled, const float* __restrict__ W1,
       const float* __restrict__ b1, const float* __restrict__ W2,
       const float* __restrict__ b2, float* __restrict__ out) {
  __shared__ float t1[NGRAPH * HCDIM];
  __shared__ float z[NGRAPH * 256];
  __shared__ float nrm[NGRAPH];
  int tid = threadIdx.x;
  for (int o = tid; o < NGRAPH * HCDIM; o += 256) {
    int r = o >> 9, j = o & 511;
    float s = b1[j];
    for (int k = 0; k < HCDIM; ++k) s += pooled[r * HCDIM + k] * W1[k * HCDIM + j];
    t1[o] = fmaxf(s, 0.f);
  }
  __syncthreads();
  for (int o = tid; o < NGRAPH * 256; o += 256) {
    int r = o >> 8, j = o & 255;
    float s = b2[j];
    for (int k = 0; k < HCDIM; ++k) s += t1[r * HCDIM + k] * W2[k * 256 + j];
    z[o] = s;
  }
  __syncthreads();
  if (tid < NGRAPH) {
    float s = 0.f;
    for (int j = 0; j < 256; ++j) { float v = z[tid * 256 + j]; s += v * v; }
    nrm[tid] = fmaxf(sqrtf(s), 1e-12f);
  }
  __syncthreads();
  for (int o = tid; o < NGRAPH * 256; o += 256) out[o] = z[o] / nrm[o >> 8];
}

// ---------------------------------------------------------------------------
// Host launcher
// ---------------------------------------------------------------------------
static inline void gemm(const unsigned short* A, const unsigned short* Bt,
                        const float* bias, float* C, unsigned short* Cbf,
                        int M, int N, int K, int ldc, hipStream_t s) {
  dim3 g(M / 32, N / 64);
  k_wmma_gemm<<<g, 256, 0, s>>>(A, Bt, bias, C, Cbf, M, N, K, ldc);
}

extern "C" void kernel_launch(void* const* d_in, const int* in_sizes, int n_in,
                              void* d_out, int out_size, void* d_ws, size_t ws_size,
                              hipStream_t stream) {
  (void)in_sizes; (void)n_in; (void)out_size; (void)ws_size;

  // Input order = setup_inputs() insertion order, recursively flattened.
  const float* x      = (const float*)d_in[0];
  const int*   ei     = (const int*)d_in[1];
  const int*   batch  = (const int*)d_in[2];
  const float* enc_W1 = (const float*)d_in[3];
  const float* enc_b1 = (const float*)d_in[4];
  const float* enc_W2 = (const float*)d_in[5];
  const float* enc_b2 = (const float*)d_in[6];
  int p = 7;
  const float *Wl[4], *bl[4], *Wr[4], *br[4], *att[4], *bias[4], *lng[4], *lnb[4];
  const float* resW = nullptr;
  for (int i = 0; i < 4; ++i) {
    Wl[i]   = (const float*)d_in[p++];
    bl[i]   = (const float*)d_in[p++];
    Wr[i]   = (const float*)d_in[p++];
    br[i]   = (const float*)d_in[p++];
    att[i]  = (const float*)d_in[p++];
    bias[i] = (const float*)d_in[p++];
    lng[i]  = (const float*)d_in[p++];
    lnb[i]  = (const float*)d_in[p++];
    if (i == 0) resW = (const float*)d_in[p++];   // only layer 0 has res_W
  }
  const float* gate_W  = (const float*)d_in[p++];
  const float* gate_b  = (const float*)d_in[p++];
  const float* proj_W1 = (const float*)d_in[p++];
  const float* proj_b1 = (const float*)d_in[p++];
  const float* proj_W2 = (const float*)d_in[p++];
  const float* proj_b2 = (const float*)d_in[p++];

  // Workspace bump allocator (256B aligned)
  size_t off = 0;
  auto alloc = [&](size_t bytes) -> void* {
    void* r = (char*)d_ws + off;
    off += (bytes + 255) & ~(size_t)255;
    return r;
  };
  float* hA            = (float*)alloc((size_t)NNODES * HCDIM * 4);
  unsigned short* hAbf = (unsigned short*)alloc((size_t)NNODES * HCDIM * 2);
  float* hB            = (float*)alloc((size_t)NNODES * HCDIM * 4);
  unsigned short* hBbf = (unsigned short*)alloc((size_t)NNODES * HCDIM * 2);
  float* xl    = (float*)alloc((size_t)NNODES * HCDIM * 4);
  float* xr    = (float*)alloc((size_t)NNODES * HCDIM * 4);
  float* resid = (float*)alloc((size_t)NNODES * HCDIM * 4);
  float* accum = (float*)alloc((size_t)NNODES * HCDIM * 4);
  float* logit = (float*)alloc((size_t)ETOT * 4 * 4);
  float* ex    = (float*)alloc((size_t)ETOT * 4 * 4);
  float* mmax  = (float*)alloc((size_t)NNODES * 4 * 4);
  float* den   = (float*)alloc((size_t)NNODES * 4 * 4);
  float* gate  = (float*)alloc((size_t)NNODES * 4);
  float* ge    = (float*)alloc((size_t)NNODES * 4);
  float* gm    = (float*)alloc(NGRAPH * 4);
  float* gden  = (float*)alloc(NGRAPH * 4);
  float* pooled = (float*)alloc((size_t)NGRAPH * HCDIM * 4);
  unsigned short* h1bf   = (unsigned short*)alloc((size_t)NNODES * 128 * 2);
  unsigned short* encW2t = (unsigned short*)alloc(128 * 128 * 2);
  unsigned short* Wlt[4], *Wrt[4];
  for (int i = 0; i < 4; ++i) {
    int inc = i ? 512 : 128;
    Wlt[i] = (unsigned short*)alloc((size_t)inc * 512 * 2);
    Wrt[i] = (unsigned short*)alloc((size_t)inc * 512 * 2);
  }
  unsigned short* resWt = (unsigned short*)alloc((size_t)128 * 512 * 2);

  // ---- weight conversion (bf16, transposed to N-major) ----
  k_w_to_bf16t<<<(128 * 128 + 255) / 256, 256, 0, stream>>>(enc_W2, encW2t, 128, 128);
  for (int i = 0; i < 4; ++i) {
    int inc = i ? 512 : 128;
    int nb = (inc * 512 + 255) / 256;
    k_w_to_bf16t<<<nb, 256, 0, stream>>>(Wl[i], Wlt[i], inc, 512);
    k_w_to_bf16t<<<nb, 256, 0, stream>>>(Wr[i], Wrt[i], inc, 512);
  }
  k_w_to_bf16t<<<(128 * 512 + 255) / 256, 256, 0, stream>>>(resW, resWt, 128, 512);

  // ---- encoder ----
  k_encoder1<<<(NNODES * 128 + 255) / 256, 256, 0, stream>>>(x, enc_W1, enc_b1, h1bf);
  gemm(h1bf, encW2t, enc_b2, hA, hAbf, NNODES, 128, 128, 128, stream);

  // ---- GATv2 layers ----
  float* cur = hA;          unsigned short* curbf = hAbf;
  float* nxt = hB;          unsigned short* nxtbf = hBbf;
  const int edgeBlocks = (ETOT + 7) / 8;        // wave per edge
  const int nodeBlocks = (NNODES + 7) / 8;      // wave per node
  for (int i = 0; i < 4; ++i) {
    int inc = i ? 512 : 128;
    gemm(curbf, Wlt[i], bl[i], xl, nullptr, NNODES, 512, inc, 512, stream);
    gemm(curbf, Wrt[i], br[i], xr, nullptr, NNODES, 512, inc, 512, stream);
    const float* residp;
    if (i == 0) {
      gemm(curbf, resWt, nullptr, resid, nullptr, NNODES, 512, 128, 512, stream);
      residp = resid;
    } else {
      residp = cur;
    }
    k_fill<<<(NNODES * 4 + 255) / 256, 256, 0, stream>>>(mmax, -INFINITY, NNODES * 4);
    k_fill<<<(NNODES * 4 + 255) / 256, 256, 0, stream>>>(den, 0.0f, NNODES * 4);
    k_fill<<<(NNODES * HCDIM + 255) / 256, 256, 0, stream>>>(accum, 0.0f, NNODES * HCDIM);
    k_edge_logit<<<edgeBlocks, 256, 0, stream>>>(ei, xl, xr, att[i], logit, mmax);
    k_edge_exp<<<(ETOT * 4 + 255) / 256, 256, 0, stream>>>(ei, logit, mmax, ex, den);
    k_edge_scatter<<<edgeBlocks, 256, 0, stream>>>(ei, xl, ex, den, accum);
    k_node_epilogue<<<nodeBlocks, 256, 0, stream>>>(accum, bias[i], lng[i], lnb[i],
                                                    residp, nxt, nxtbf);
    float* tf = cur; cur = nxt; nxt = tf;
    unsigned short* tb = curbf; curbf = nxtbf; nxtbf = tb;
  }

  // ---- gate pooling ----
  k_fill<<<1, 256, 0, stream>>>(gm, -INFINITY, NGRAPH);
  k_fill<<<1, 256, 0, stream>>>(gden, 0.0f, NGRAPH);
  k_fill<<<(NGRAPH * HCDIM + 255) / 256, 256, 0, stream>>>(pooled, 0.0f, NGRAPH * HCDIM);
  k_gate_logit<<<nodeBlocks, 256, 0, stream>>>(cur, gate_W, gate_b, batch, gate, gm);
  k_gate_exp<<<(NNODES + 255) / 256, 256, 0, stream>>>(gate, gm, batch, ge, gden);
  k_gate_pool<<<nodeBlocks, 256, 0, stream>>>(cur, ge, gden, batch, pooled);

  // ---- projection + normalize ----
  k_proj<<<1, 256, 0, stream>>>(pooled, proj_W1, proj_b1, proj_W2, proj_b2,
                                (float*)d_out);
}
